// ContrastiveLoss_59803124629833
// MI455X (gfx1250) — matrix operand order repaired and checked by reading, not compiled, not run on hardware
//
#include <hip/hip_runtime.h>
#include <hip/hip_bf16.h>

typedef __attribute__((ext_vector_type(8)))  __bf16 v8bf;
typedef __attribute__((ext_vector_type(16))) __bf16 v16bf;
typedef __attribute__((ext_vector_type(8)))  float  v8f;

#define BATCH 8
#define CH    256
#define HI    128
#define WI    256
#define OH    32
#define OW    64
#define NPOS  2048          // OH*OW
#define IGNORE_LAB 19
#define INV_TAU 14.285714285714286f
#define LOG2E  1.4426950408889634f

__device__ __forceinline__ unsigned short f2bf(float x) {
  unsigned int u = __float_as_uint(x);
  u += 0x7FFFu + ((u >> 16) & 1u);     // round-to-nearest-even
  return (unsigned short)(u >> 16);
}

// ---------------------------------------------------------------------------
// Kernel 0: zero the 16 global accumulators (llsum[8], validcnt[8])
// ---------------------------------------------------------------------------
__global__ void init_acc_kernel(float* __restrict__ acc) {
  if (threadIdx.x < 16) acc[threadIdx.x] = 0.0f;
}

// ---------------------------------------------------------------------------
// Kernel 1: bilinear 4x downsample (exact 2x2 average for this geometry),
// L2-normalize each 256-d feature row, emit bf16 g[b][n][c]; nearest labels.
// Feature taps use non-temporal loads (256 MB single-use stream) so the 8 MB
// g matrix stays L2-resident for kernel 2's re-reads.
// ---------------------------------------------------------------------------
__global__ void __launch_bounds__(256) resize_norm_kernel(
    const float* __restrict__ feat, const int* __restrict__ lab,
    unsigned short* __restrict__ g, int* __restrict__ labw)
{
  __shared__ float vbuf[CH * 65];   // padded stride 65 to kill bank conflicts
  __shared__ float ssq4[4 * 64];
  __shared__ float inv64[64];

  const int b  = blockIdx.x >> 5;
  const int oh = blockIdx.x & 31;
  const int t  = threadIdx.x;
  const int j  = t & 63;            // output column 0..63
  const int cg = t >> 6;            // channel group 0..3
  const int h0 = 4 * oh + 1;        // bilinear taps: rows h0, h0+1
  const int w0 = 4 * j + 1;         // cols w0, w0+1

  const float* fb = feat + (((size_t)(b * CH)) * HI + h0) * WI + w0;
  float ssq = 0.0f;
  for (int c = cg * 64; c < cg * 64 + 64; ++c) {
    const float* p = fb + (size_t)c * (HI * WI);
    float v = 0.25f * (__builtin_nontemporal_load(p) +
                       __builtin_nontemporal_load(p + 1) +
                       __builtin_nontemporal_load(p + WI) +
                       __builtin_nontemporal_load(p + WI + 1));
    vbuf[c * 65 + j] = v;
    ssq += v * v;
  }
  ssq4[cg * 64 + j] = ssq;
  __syncthreads();

  if (t < 64) {
    float s = ssq4[t] + ssq4[64 + t] + ssq4[128 + t] + ssq4[192 + t];
    inv64[t] = 1.0f / fmaxf(sqrtf(s), 1e-12f);
    labw[b * NPOS + oh * OW + t] = lab[(b * HI + 4 * oh) * WI + 4 * t];
  }
  __syncthreads();

  // Coalesced bf16 writes: 2 rows per pass, 128 threads per row (1 dword each)
  const int rhalf = t >> 7;         // 0..1
  const int cpair = t & 127;        // dword index in row
  for (int jj = 0; jj < 64; jj += 2) {
    const int row = jj + rhalf;
    const float inv = inv64[row];
    const int c = cpair * 2;
    unsigned int lo = f2bf(vbuf[c * 65 + row] * inv);
    unsigned int hi = f2bf(vbuf[(c + 1) * 65 + row] * inv);
    unsigned int* dst =
        (unsigned int*)(g + ((size_t)(b * NPOS + oh * OW + row)) * CH);
    dst[cpair] = lo | (hi << 16);
  }
}

// ---------------------------------------------------------------------------
// Kernel 2 helpers
// ---------------------------------------------------------------------------
// Load one 32x16 B tile (two 16B loads per K-chunk, contiguous K order, same as A:
// lane holds 16 contiguous K of column n = lane%16; lanes>=16 carry K 16..31).
__device__ __forceinline__ void load_btile(const unsigned short* __restrict__ brow,
                                           v16bf bF[8]) {
#pragma unroll
  for (int kc = 0; kc < 8; ++kc) {
    const v8bf b0 = *(const v8bf*)(brow + kc * 32);
    const v8bf b1 = *(const v8bf*)(brow + kc * 32 + 8);
    bF[kc] = __builtin_shufflevector(b0, b1, 0, 1, 2, 3, 4, 5, 6, 7,
                                             8, 9, 10, 11, 12, 13, 14, 15);
  }
}

// 16x16 tile over K=256: two independent 4-WMMA chains (intra-wave XDL ILP),
// then fold into per-row statistics.
__device__ __forceinline__ void tile_stats(const v16bf aF[8], const v16bf bF[8],
                                           int labn, const int labm[8],
                                           float se[8], float sm[8], float ct[8]) {
  v8f acc0 = {};
  v8f acc1 = {};
#pragma unroll
  for (int kc = 0; kc < 4; ++kc) {
    acc0 = __builtin_amdgcn_wmma_f32_16x16x32_bf16(
        false, aF[kc], false, bF[kc], (short)0, acc0, false, false);
    acc1 = __builtin_amdgcn_wmma_f32_16x16x32_bf16(
        false, aF[kc + 4], false, bF[kc + 4], (short)0, acc1, false, false);
  }
#pragma unroll
  for (int r = 0; r < 8; ++r) {
    const float dot = acc0[r] + acc1[r];
    // exp(dot/tau) == exp2(dot * log2e/tau); |arg| <= ~21 so the raw hardware
    // v_exp_f32 (no denormal-range guard) is safe.
    se[r] += __builtin_amdgcn_exp2f(dot * (INV_TAU * LOG2E));
    const bool w = (labn == labm[r]) && (labm[r] != IGNORE_LAB);
    sm[r] += w ? dot : 0.0f;         // raw dot; *INV_TAU applied at the end
    ct[r] += w ? 1.0f : 0.0f;
  }
}

// ---------------------------------------------------------------------------
// Kernel 2: fused  logits = (G G^T)/tau  + per-row {sum exp, masked sum, count}
// One WG per (batch, 16-row tile); 8 waves sweep the 128 column tiles with
// double-buffered B operands so L2 latency hides behind the WMMA pipe.
// ---------------------------------------------------------------------------
__global__ void __launch_bounds__(256) gemm_stats_kernel(
    const unsigned short* __restrict__ g, const int* __restrict__ labw,
    float* __restrict__ llsum, float* __restrict__ vcnt)
{
  __shared__ float redSe[16];
  __shared__ float redSm[16];
  __shared__ float redCt[16];

  const int b     = blockIdx.x >> 7;
  const int mtile = blockIdx.x & 127;
  const int m0    = mtile * 16;
  const int tid   = threadIdx.x;
  const int wave  = tid >> 5;
  const int lane  = tid & 31;
  const int lhalf = lane >> 4;      // 0: lanes 0-15, 1: lanes 16-31
  const int l16   = lane & 15;

  if (tid < 16) { redSe[tid] = 0.0f; redSm[tid] = 0.0f; redCt[tid] = 0.0f; }
  __syncthreads();

  const unsigned short* gb   = g + (size_t)b * NPOS * CH;
  const int*            labb = labw + b * NPOS;

  // --- A fragments: rows m0..m0+15, full K=256, resident in VGPRs.
  // A in contiguous K order (16x32): lanes 0-15 hold K 0..15 of row l16,
  // lanes 16-31 hold K 16..31.
  v16bf aF[8];
  {
    const unsigned short* arow = gb + (size_t)(m0 + l16) * CH + lhalf * 16;
#pragma unroll
    for (int kc = 0; kc < 8; ++kc) {
      const v8bf a0 = *(const v8bf*)(arow + kc * 32);
      const v8bf a1 = *(const v8bf*)(arow + kc * 32 + 8);
      aF[kc] = __builtin_shufflevector(a0, a1, 0, 1, 2, 3, 4, 5, 6, 7,
                                               8, 9, 10, 11, 12, 13, 14, 15);
    }
  }
  // C/D layout: element (M = r + 8*lhalf, N = l16) lives in acc reg r.
  int labm[8];
#pragma unroll
  for (int r = 0; r < 8; ++r) labm[r] = labb[m0 + r + 8 * lhalf];

  float se[8], sm[8], ct[8];
#pragma unroll
  for (int r = 0; r < 8; ++r) { se[r] = 0.0f; sm[r] = 0.0f; ct[r] = 0.0f; }

  // Double-buffered sweep: each wave handles tiles wave, wave+8, ..., 16 total.
  v16bf bPing[8], bPong[8];
  #define BROW(tt) (gb + (size_t)((tt) * 16 + l16) * CH + lhalf * 16)

  load_btile(BROW(wave), bPing);
#pragma unroll 1
  for (int t = wave; t < 128; t += 16) {
    load_btile(BROW(t + 8), bPong);            // prefetch next tile
    const int labn0 = labb[t * 16 + l16];
    tile_stats(aF, bPing, labn0, labm, se, sm, ct);

    if (t + 16 < 128) load_btile(BROW(t + 16), bPing);  // prefetch tile after
    const int labn1 = labb[(t + 8) * 16 + l16];
    tile_stats(aF, bPong, labn1, labm, se, sm, ct);
  }
  #undef BROW

  // Reduce over the 16 lanes of each half (N dimension), xor masks 1,2,4,8
#pragma unroll
  for (int off = 1; off < 16; off <<= 1) {
#pragma unroll
    for (int r = 0; r < 8; ++r) {
      se[r] += __shfl_xor(se[r], off, 32);
      sm[r] += __shfl_xor(sm[r], off, 32);
      ct[r] += __shfl_xor(ct[r], off, 32);
    }
  }
  if (l16 == 0) {  // lanes 0 (rows 0-7) and 16 (rows 8-15)
#pragma unroll
    for (int r = 0; r < 8; ++r) {
      const int row = r + 8 * lhalf;
      atomicAdd(&redSe[row], se[r]);   // ds_add_f32
      atomicAdd(&redSm[row], sm[r]);
      atomicAdd(&redCt[row], ct[r]);
    }
  }
  __syncthreads();

  if (tid < 16) {
    const float den   = redCt[tid];
    const float valid = (den > 0.0f) ? 1.0f : 0.0f;
    const float d     = (den > 0.0f) ? den : 1.0f;
    // ll = LSE - S/den ;  LSE = log(sum exp2(dot*log2e/tau)) = log(sum exp(dot/tau))
    const float ll = valid * (__logf(redSe[tid]) - (redSm[tid] * INV_TAU) / d);
    atomicAdd(&llsum[b], ll);          // global_atomic_add_f32
    atomicAdd(&vcnt[b], valid);
  }
}

// ---------------------------------------------------------------------------
// Kernel 3: loss = mean_b( llsum[b] / max(cnt[b],1) )
// ---------------------------------------------------------------------------
__global__ void final_kernel(const float* __restrict__ llsum,
                             const float* __restrict__ vcnt,
                             float* __restrict__ out)
{
  if (threadIdx.x == 0) {
    float acc = 0.0f;
    for (int b = 0; b < BATCH; ++b) {
      const float c = vcnt[b];
      acc += llsum[b] / ((c > 0.0f) ? c : 1.0f);
    }
    out[0] = acc * (1.0f / BATCH);
  }
}

// ---------------------------------------------------------------------------
extern "C" void kernel_launch(void* const* d_in, const int* in_sizes, int n_in,
                              void* d_out, int out_size, void* d_ws, size_t ws_size,
                              hipStream_t stream) {
  (void)in_sizes; (void)n_in; (void)out_size; (void)ws_size;
  const float* feat = (const float*)d_in[0];
  const int*   lab  = (const int*)d_in[1];
  float*       out  = (float*)d_out;

  // Workspace layout
  unsigned short* g = (unsigned short*)d_ws;                       // 8 MB bf16
  char* p = (char*)d_ws + (size_t)BATCH * NPOS * CH * sizeof(unsigned short);
  int*   labw  = (int*)p;                                          // 64 KB
  float* accum = (float*)(p + (size_t)BATCH * NPOS * sizeof(int)); // 16 floats
  float* llsum = accum;
  float* vcnt  = accum + 8;

  init_acc_kernel<<<1, 32, 0, stream>>>(accum);
  resize_norm_kernel<<<BATCH * OH, 256, 0, stream>>>(feat, lab, g, labw);
  gemm_stats_kernel<<<BATCH * (NPOS / 16), 256, 0, stream>>>(g, labw, llsum, vcnt);
  final_kernel<<<1, 32, 0, stream>>>(llsum, vcnt, out);
}
